// SpatioTemporal_Vision_Transformer_12953621365116
// MI455X (gfx1250) — matrix operand order repaired and checked by reading, compiled, not run on hardware
//
#include <hip/hip_runtime.h>

typedef __attribute__((ext_vector_type(16))) _Float16 v16h;
typedef __attribute__((ext_vector_type(8)))  _Float16 v8h;
typedef __attribute__((ext_vector_type(8)))  float    v8f;

namespace vit {
constexpr int IMG = 224, PAT = 16, CH = 3, D = 768, NHEADS = 12, HDIM = 64;
constexpr int NL = 6, HIDN = 3072, OUTD = 2;
constexpr int G = 14, NPAT = 196, SEQ = 197, B = 8;
constexpr int ROWS_H = B * SEQ;    // 1576
constexpr int ROWS_S = NPAT * B;   // 1568
// workspace offsets (floats)
constexpr long O_PATCH = 0;                                      // 1568*768
constexpr long O_H     = O_PATCH + (long)ROWS_S * D;             // 1576*768
constexpr long O_TMP   = O_H     + (long)ROWS_H * D;             // 1576*768
constexpr long O_QKV   = O_TMP   + (long)ROWS_H * D;             // 1576*2304
constexpr long O_SC    = O_QKV   + (long)ROWS_H * 3 * D;         // 8*12*197*197
constexpr long O_AO    = O_SC    + (long)B * NHEADS * SEQ * SEQ; // 1576*768
constexpr long O_MLPH  = O_AO    + (long)ROWS_H * D;             // 1576*3072
constexpr long O_ORIX  = O_MLPH  + (long)ROWS_H * HIDN;          // 8*768
// aliases (lifetimes disjoint)
constexpr long O_XM  = O_PATCH;   // 8*768*196
constexpr long O_X1  = O_QKV;     // 196*8*768
constexpr long O_V   = O_SC;      // 1568*768
constexpr long O_XS  = O_AO;      // 1568*768
constexpr long O_HID = O_MLPH;    // 1568*3072
}

// ---------------------------------------------------------------------------
// Generic strided, batched, LDS-staged WMMA GEMM with fused epilogue.
//   C[r, n] = act( scale * sum_k A[r,k]*B[k,n] + bias[n] + R[r,n] )
// A element at A[r*rsA + k*csA] (+ z offsets), B at B[k*rsB + n*csB].
// All intra-matrix offsets are 32-bit (matrices < 2^31 elements) so loads
// lower to SADDR + 32-bit VGPR-offset form (no 64-bit address math).
// Block = 128 threads = 4 waves; tile 64x64, K-step 32.
// Main K-loop is guard-free (per-thread base offsets hoisted); ragged K
// handled by one fully-guarded tail step. Staging is branchless
// (clamped-address loads + value select).
// ---------------------------------------------------------------------------
struct GemmArgs {
  const float* A; const float* B; const float* bias; const float* R; float* C;
  int M, N, K;
  int rsA, csA, rsB, csB, rsC, rsR, csR;
  int nh, nz;
  long sAb, sAh, sBb, sBh, sCb, sCh, sRb, sRh;
  int act;        // 0=none, 1=gelu(exact), 2=tanh
  float scale;
};

__global__ __launch_bounds__(128) void gemm_wmma_k(GemmArgs g) {
  __shared__ __align__(16) _Float16 Ah[64 * 32];   // [row][k], k contiguous
  __shared__ __align__(16) _Float16 Bt[64 * 32];   // [n][k],  k contiguous (transposed)

  const int z  = blockIdx.z;
  const int zb = z / g.nh, zh = z % g.nh;
  const float* Ap = g.A + (long)zb * g.sAb + (long)zh * g.sAh;
  const float* Bp = g.B + (long)zb * g.sBb + (long)zh * g.sBh;
  float*       Cp = g.C + (long)zb * g.sCb + (long)zh * g.sCh;
  const float* Rp = g.R ? (g.R + (long)zb * g.sRb + (long)zh * g.sRh) : nullptr;

  const int tid  = threadIdx.x;
  const int lane = tid & 31;
  const int wave = tid >> 5;
  const int lm   = lane & 15;
  const bool hi  = lane >= 16;
  const int rowBase = blockIdx.y * 64;
  const int colBase = blockIdx.x * 64;

  v8f acc[4];
#pragma unroll
  for (int t = 0; t < 4; ++t) acc[t] = (v8f){0.f,0.f,0.f,0.f,0.f,0.f,0.f,0.f};

  // ---- per-thread staging geometry (loop-invariant) ----
  // A: pair p = i*128+tid -> row r = p/16, k = 2*(p%16). LDS word idx == p.
  unsigned aOff[8]; bool aOk[8];
#pragma unroll
  for (int i = 0; i < 8; ++i) {
    const int p = i * 128 + tid;
    const int r = p >> 4, kp = p & 15;
    const int gr = rowBase + r;
    const int cr = gr < g.M ? gr : g.M - 1;
    aOk[i]  = gr < g.M;
    aOff[i] = (unsigned)cr * (unsigned)g.rsA + (unsigned)(2 * kp) * (unsigned)g.csA;
  }
  // B: pair p = i*128+tid -> n = p%64, k = 2*(p/64). LDS word idx = n*16 + p/64.
  unsigned bOff[8]; bool bOk[8]; int bWord[8];
#pragma unroll
  for (int i = 0; i < 8; ++i) {
    const int p = i * 128 + tid;
    const int n = p & 63, kp = p >> 6;
    const int gn = colBase + n;
    const int cn = gn < g.N ? gn : g.N - 1;
    bOk[i]   = gn < g.N;
    bOff[i]  = (unsigned)(2 * kp) * (unsigned)g.rsB + (unsigned)cn * (unsigned)g.csB;
    bWord[i] = n * 16 + kp;
  }
  unsigned* const AhW = (unsigned*)Ah;
  unsigned* const BtW = (unsigned*)Bt;

  // ---- fragment compute from staged LDS tiles: all DS loads, then 4 WMMAs ----
  const int arow  = wave * 16 + lm;
  const int kaOff = hi ? 8 : 0;
  const int kbOff = hi ? 8 : 0;
  auto compute = [&]() {
    const v8h a0 = *(const v8h*)&Ah[arow * 32 + kaOff];
    const v8h a1 = *(const v8h*)&Ah[arow * 32 + kaOff + 16];
    v8h bfr[8];
#pragma unroll
    for (int t = 0; t < 4; ++t) {
      const int n = t * 16 + lm;
      bfr[2 * t]     = *(const v8h*)&Bt[n * 32 + kbOff];
      bfr[2 * t + 1] = *(const v8h*)&Bt[n * 32 + kbOff + 16];
    }
    const v16h a = __builtin_shufflevector(a0, a1,
        0, 1, 2, 3, 4, 5, 6, 7, 8, 9, 10, 11, 12, 13, 14, 15);
#pragma unroll
    for (int t = 0; t < 4; ++t) {
      const v16h bf = __builtin_shufflevector(bfr[2 * t], bfr[2 * t + 1],
          0, 1, 2, 3, 4, 5, 6, 7, 8, 9, 10, 11, 12, 13, 14, 15);
      acc[t] = __builtin_amdgcn_wmma_f32_16x16x32_f16(
          false, a, false, bf, (short)0, acc[t], false, false);
    }
  };

  auto pack2 = [](_Float16 h0, _Float16 h1) -> unsigned {
    return (unsigned)__builtin_bit_cast(unsigned short, h0) |
           ((unsigned)__builtin_bit_cast(unsigned short, h1) << 16);
  };

  const int Kmain = g.K & ~31;
  for (int k0 = 0; k0 < Kmain; k0 += 32) {
    __syncthreads();   // protect LDS reads of previous iteration
    const unsigned kOffA = (unsigned)k0 * (unsigned)g.csA;   // uniform
    const unsigned kOffB = (unsigned)k0 * (unsigned)g.rsB;   // uniform
    float a0v[8], a1v[8], b0v[8], b1v[8];
#pragma unroll
    for (int i = 0; i < 8; ++i) {
      const unsigned o = aOff[i] + kOffA;
      a0v[i] = Ap[o];
      a1v[i] = Ap[o + (unsigned)g.csA];
    }
#pragma unroll
    for (int i = 0; i < 8; ++i) {
      const unsigned o = bOff[i] + kOffB;
      b0v[i] = Bp[o];
      b1v[i] = Bp[o + (unsigned)g.rsB];
    }
    if (k0 + 32 < g.K) {   // uniform prefetch of next K tile
      __builtin_prefetch(Ap + (aOff[0] + kOffA + 32u * (unsigned)g.csA), 0, 3);
      __builtin_prefetch(Bp + (bOff[0] + kOffB + 32u * (unsigned)g.rsB), 0, 3);
    }
#pragma unroll
    for (int i = 0; i < 8; ++i) {
      const _Float16 h0 = (_Float16)(aOk[i] ? a0v[i] : 0.f);
      const _Float16 h1 = (_Float16)(aOk[i] ? a1v[i] : 0.f);
      AhW[i * 128 + tid] = pack2(h0, h1);
    }
#pragma unroll
    for (int i = 0; i < 8; ++i) {
      const _Float16 h0 = (_Float16)(bOk[i] ? b0v[i] : 0.f);
      const _Float16 h1 = (_Float16)(bOk[i] ? b1v[i] : 0.f);
      BtW[bWord[i]] = pack2(h0, h1);
    }
    __syncthreads();
    compute();
  }

  // ---- guarded tail step (K % 32 != 0) ----
  if (Kmain < g.K) {
    __syncthreads();
    float va[16], vb[16];
#pragma unroll
    for (int i = 0; i < 16; ++i) {
      const int e = i * 128 + tid;
      const int r = e >> 5, k = e & 31;
      const int gr = rowBase + r, gk = Kmain + k;
      const int cr = gr < g.M ? gr : g.M - 1;
      const int ck = gk < g.K ? gk : g.K - 1;
      va[i] = Ap[(unsigned)cr * (unsigned)g.rsA + (unsigned)ck * (unsigned)g.csA];
    }
#pragma unroll
    for (int i = 0; i < 16; ++i) {
      const int e = i * 128 + tid;
      const int k = e >> 6, n = e & 63;
      const int gk = Kmain + k, gn = colBase + n;
      const int ck = gk < g.K ? gk : g.K - 1;
      const int cn = gn < g.N ? gn : g.N - 1;
      vb[i] = Bp[(unsigned)ck * (unsigned)g.rsB + (unsigned)cn * (unsigned)g.csB];
    }
#pragma unroll
    for (int i = 0; i < 16; ++i) {
      const int e = i * 128 + tid;
      const int r = e >> 5, k = e & 31;
      const bool ok = (rowBase + r < g.M) && (Kmain + k < g.K);
      Ah[e] = (_Float16)(ok ? va[i] : 0.f);
    }
#pragma unroll
    for (int i = 0; i < 16; ++i) {
      const int e = i * 128 + tid;
      const int k = e >> 6, n = e & 63;
      const bool ok = (Kmain + k < g.K) && (colBase + n < g.N);
      Bt[n * 32 + k] = (_Float16)(ok ? vb[i] : 0.f);
    }
    __syncthreads();
    compute();
  }

  // ---- fused epilogue ----
#pragma unroll
  for (int t = 0; t < 4; ++t) {
    const int n = colBase + t * 16 + lm;
    if (n >= g.N) continue;
    const float bv = g.bias ? g.bias[n] : 0.f;
#pragma unroll
    for (int v = 0; v < 8; ++v) {
      const int r = rowBase + wave * 16 + v + (hi ? 8 : 0);
      if (r >= g.M) continue;
      float val = acc[t][v] * g.scale + bv;
      if (Rp) val += Rp[(unsigned)r * (unsigned)g.rsR + (unsigned)n * (unsigned)g.csR];
      if (g.act == 1) val = 0.5f * val * (1.f + erff(val * 0.70710678118654752f));
      else if (g.act == 2) val = tanhf(val);
      Cp[(unsigned)r * (unsigned)g.rsC + (unsigned)n] = val;
    }
  }
}

// ---------------------------------------------------------------------------
// Elementwise / reduction kernels
// ---------------------------------------------------------------------------
__global__ __launch_bounds__(256) void patchify_k(const float* __restrict__ x,
                                                  float* __restrict__ p) {
  using namespace vit;
  const long NT = (long)B * NPAT * D;
  long idx = (long)blockIdx.x * 256 + threadIdx.x;
  if (idx >= NT) return;
  int f  = (int)(idx % D);
  int pi = (int)((idx / D) % NPAT);
  int b  = (int)(idx / ((long)D * NPAT));
  int c  = f % CH;
  int px = (f / CH) % PAT;
  int py = f / (CH * PAT);
  int gy = pi / G, gx = pi % G;
  p[idx] = x[(((long)b * CH + c) * IMG + (gy * PAT + py)) * IMG + gx * PAT + px];
}

__global__ __launch_bounds__(256) void build_h_k(const float* __restrict__ enc,
                                                 const float* __restrict__ cls,
                                                 const float* __restrict__ pos,
                                                 float* __restrict__ h) {
  using namespace vit;
  const long NT = (long)B * SEQ * D;
  long idx = (long)blockIdx.x * 256 + threadIdx.x;
  if (idx >= NT) return;
  int d = (int)(idx % D);
  int s = (int)((idx / D) % SEQ);
  int b = (int)(idx / ((long)D * SEQ));
  float v = (s == 0) ? cls[d] : enc[((long)b * NPAT + (s - 1)) * D + d];
  h[idx] = v + pos[(long)s * D + d];
}

__global__ __launch_bounds__(256) void layernorm_k(const float* __restrict__ x,
                                                   const float* __restrict__ gam,
                                                   const float* __restrict__ bet,
                                                   float* __restrict__ y, int Dd) {
  __shared__ float red[256];
  const long row = blockIdx.x;
  const float* xr = x + row * (long)Dd;
  float s = 0.f;
  for (int i = threadIdx.x; i < Dd; i += 256) s += xr[i];
  red[threadIdx.x] = s; __syncthreads();
  for (int o = 128; o > 0; o >>= 1) {
    if ((int)threadIdx.x < o) red[threadIdx.x] += red[threadIdx.x + o];
    __syncthreads();
  }
  const float mu = red[0] / Dd; __syncthreads();
  float v = 0.f;
  for (int i = threadIdx.x; i < Dd; i += 256) { float d = xr[i] - mu; v += d * d; }
  red[threadIdx.x] = v; __syncthreads();
  for (int o = 128; o > 0; o >>= 1) {
    if ((int)threadIdx.x < o) red[threadIdx.x] += red[threadIdx.x + o];
    __syncthreads();
  }
  const float rstd = rsqrtf(red[0] / Dd + 1e-5f);
  for (int i = threadIdx.x; i < Dd; i += 256)
    y[row * (long)Dd + i] = (xr[i] - mu) * rstd * gam[i] + bet[i];
}

__global__ __launch_bounds__(256) void softmax_k(float* __restrict__ x, int L) {
  __shared__ float red[256];
  const long row = blockIdx.x;
  float* xr = x + row * (long)L;
  float mx = -3.0e38f;
  for (int i = threadIdx.x; i < L; i += 256) mx = fmaxf(mx, xr[i]);
  red[threadIdx.x] = mx; __syncthreads();
  for (int o = 128; o > 0; o >>= 1) {
    if ((int)threadIdx.x < o) red[threadIdx.x] = fmaxf(red[threadIdx.x], red[threadIdx.x + o]);
    __syncthreads();
  }
  mx = red[0]; __syncthreads();
  float s = 0.f;
  for (int i = threadIdx.x; i < L; i += 256) { float e = expf(xr[i] - mx); xr[i] = e; s += e; }
  red[threadIdx.x] = s; __syncthreads();
  for (int o = 128; o > 0; o >>= 1) {
    if ((int)threadIdx.x < o) red[threadIdx.x] += red[threadIdx.x + o];
    __syncthreads();
  }
  const float inv = 1.f / red[0];
  for (int i = threadIdx.x; i < L; i += 256) xr[i] *= inv;
}

// orix1[b*768+d] = sum_j xm[b,d,j]*w[j] + bias[0]
__global__ __launch_bounds__(256) void mlp_single_k(const float* __restrict__ xm,
                                                    const float* __restrict__ w,
                                                    const float* __restrict__ bias,
                                                    float* __restrict__ o) {
  using namespace vit;
  int idx = blockIdx.x * 256 + threadIdx.x;
  if (idx >= B * D) return;
  const float* r = xm + (long)idx * NPAT;
  float s = 0.f;
  for (int j = 0; j < NPAT; ++j) s += r[j] * w[j];
  o[idx] = s + bias[0];
}

// ---------------------------------------------------------------------------
// Host side
// ---------------------------------------------------------------------------
static GemmArgs mk(const float* A, const float* Bm, const float* bias, const float* R,
                   float* C, int M, int N, int K,
                   int rsA, int csA, int rsB, int csB, int rsC) {
  GemmArgs g{};
  g.A = A; g.B = Bm; g.bias = bias; g.R = R; g.C = C;
  g.M = M; g.N = N; g.K = K;
  g.rsA = rsA; g.csA = csA; g.rsB = rsB; g.csB = csB; g.rsC = rsC;
  g.rsR = rsC; g.csR = 1;
  g.nh = 1; g.nz = 1;
  g.sAb = g.sAh = g.sBb = g.sBh = g.sCb = g.sCh = g.sRb = g.sRh = 0;
  g.act = 0; g.scale = 1.f;
  return g;
}

static void run_gemm(hipStream_t st, const GemmArgs& g) {
  dim3 grid((g.N + 63) / 64, (g.M + 63) / 64, g.nz);
  gemm_wmma_k<<<grid, dim3(128), 0, st>>>(g);
}

extern "C" void kernel_launch(void* const* d_in, const int* in_sizes, int n_in,
                              void* d_out, int out_size, void* d_ws, size_t ws_size,
                              hipStream_t stream) {
  using namespace vit;
  (void)in_sizes; (void)n_in; (void)out_size; (void)ws_size;
  const float* x       = (const float*)d_in[0];
  const float* enc_w   = (const float*)d_in[1];
  const float* enc_b   = (const float*)d_in[2];
  const float* cls     = (const float*)d_in[3];
  const float* pos     = (const float*)d_in[4];
  // layers: 5 + 11*l : ln1.g, ln1.b, qkv.w, proj.w, proj.b, ln2.g, ln2.b,
  //                    fc1.w, fc1.b, fc2.w, fc2.b
  const int LBASE = 5;
  const float* mlpm_w  = (const float*)d_in[71];
  const float* mlpm_b  = (const float*)d_in[72];
  const float* mlps_w  = (const float*)d_in[73];
  const float* mlps_b  = (const float*)d_in[74];
  const float* sfl_w   = (const float*)d_in[75];
  const float* sfl_b   = (const float*)d_in[76];
  const float* sn_g    = (const float*)d_in[77];
  const float* sn_b    = (const float*)d_in[78];
  const float* sqkv_w  = (const float*)d_in[79];
  const float* sproj_w = (const float*)d_in[80];
  const float* sproj_b = (const float*)d_in[81];
  const float* phln_g  = (const float*)d_in[82];
  const float* phln_b  = (const float*)d_in[83];
  const float* hf1_w   = (const float*)d_in[84];
  const float* hf1_b   = (const float*)d_in[85];
  const float* hf2_w   = (const float*)d_in[86];
  const float* hf2_b   = (const float*)d_in[87];
  float* out = (float*)d_out;

  float* ws      = (float*)d_ws;
  float* patches = ws + O_PATCH;
  float* h       = ws + O_H;
  float* tmp     = ws + O_TMP;
  float* qkv     = ws + O_QKV;
  float* sc      = ws + O_SC;
  float* ao      = ws + O_AO;
  float* mlph    = ws + O_MLPH;
  float* orix    = ws + O_ORIX;
  float* xm      = ws + O_XM;
  float* x1      = ws + O_X1;
  float* vbuf    = ws + O_V;
  float* xs      = ws + O_XS;
  float* hid     = ws + O_HID;

  // 1) patchify: (8,3,224,224) -> (1568, 768)
  {
    long nt = (long)B * NPAT * D;
    patchify_k<<<dim3((unsigned)((nt + 255) / 256)), dim3(256), 0, stream>>>(x, patches);
  }
  // 2) enc = patches @ lin_enc.w + b -> tmp (1568 x 768)
  run_gemm(stream, mk(patches, enc_w, enc_b, nullptr, tmp,
                      ROWS_S, D, D, D, 1, D, 1, D));
  // 3) h = concat(cls, enc) + pos  -> (8,197,768)
  {
    long nt = (long)B * SEQ * D;
    build_h_k<<<dim3((unsigned)((nt + 255) / 256)), dim3(256), 0, stream>>>(tmp, cls, pos, h);
  }
  // 4) transformer layers
  for (int l = 0; l < NL; ++l) {
    const float* ln1g = (const float*)d_in[LBASE + 11 * l + 0];
    const float* ln1b = (const float*)d_in[LBASE + 11 * l + 1];
    const float* qkvw = (const float*)d_in[LBASE + 11 * l + 2];
    const float* prjw = (const float*)d_in[LBASE + 11 * l + 3];
    const float* prjb = (const float*)d_in[LBASE + 11 * l + 4];
    const float* ln2g = (const float*)d_in[LBASE + 11 * l + 5];
    const float* ln2b = (const float*)d_in[LBASE + 11 * l + 6];
    const float* f1w  = (const float*)d_in[LBASE + 11 * l + 7];
    const float* f1b  = (const float*)d_in[LBASE + 11 * l + 8];
    const float* f2w  = (const float*)d_in[LBASE + 11 * l + 9];
    const float* f2b  = (const float*)d_in[LBASE + 11 * l + 10];

    layernorm_k<<<dim3(ROWS_H), dim3(256), 0, stream>>>(h, ln1g, ln1b, tmp, D);
    // qkv = ln(h) @ Wqkv  (1576 x 2304)
    run_gemm(stream, mk(tmp, qkvw, nullptr, nullptr, qkv,
                        ROWS_H, 3 * D, D, 3 * D, 1, 3 * D, 1, 3 * D));
    // scores[b,hd,i,j] = (q_i . k_j) / 8  — batched over z = b*12+hd
    {
      GemmArgs g = mk(qkv, qkv + D, nullptr, nullptr, sc,
                      SEQ, SEQ, HDIM,
                      /*rsA*/ 3 * D, /*csA*/ 1,
                      /*rsB (k^T)*/ 1, /*csB*/ 3 * D,
                      /*rsC*/ SEQ);
      g.nh = NHEADS; g.nz = B * NHEADS;
      g.sAb = (long)SEQ * 3 * D; g.sAh = HDIM;
      g.sBb = (long)SEQ * 3 * D; g.sBh = HDIM;
      g.sCb = (long)NHEADS * SEQ * SEQ; g.sCh = (long)SEQ * SEQ;
      g.scale = 0.125f;  // HD^-0.5
      run_gemm(stream, g);
    }
    softmax_k<<<dim3(B * NHEADS * SEQ), dim3(256), 0, stream>>>(sc, SEQ);
    // ao[b,i,hd*64+n] = sum_j a[i,j] * v[j,n]
    {
      GemmArgs g = mk(sc, qkv + 2 * D, nullptr, nullptr, ao,
                      SEQ, HDIM, SEQ,
                      /*rsA*/ SEQ, /*csA*/ 1,
                      /*rsB*/ 3 * D, /*csB*/ 1,
                      /*rsC*/ D);
      g.nh = NHEADS; g.nz = B * NHEADS;
      g.sAb = (long)NHEADS * SEQ * SEQ; g.sAh = (long)SEQ * SEQ;
      g.sBb = (long)SEQ * 3 * D; g.sBh = HDIM;
      g.sCb = (long)SEQ * D; g.sCh = HDIM;
      run_gemm(stream, g);
    }
    // h = h + ao @ Wproj + bproj   (residual fused, in-place safe)
    run_gemm(stream, mk(ao, prjw, prjb, h, h, ROWS_H, D, D, D, 1, D, 1, D));
    // MLP
    layernorm_k<<<dim3(ROWS_H), dim3(256), 0, stream>>>(h, ln2g, ln2b, tmp, D);
    {
      GemmArgs g = mk(tmp, f1w, f1b, nullptr, mlph,
                      ROWS_H, HIDN, D, D, 1, HIDN, 1, HIDN);
      g.act = 1;  // exact gelu
      run_gemm(stream, g);
    }
    run_gemm(stream, mk(mlph, f2w, f2b, h, h, ROWS_H, D, HIDN, HIDN, 1, D, 1, D));
  }
  // 5) xm[b,d,j] = sum_s h[b,s,d] * Wm[s,j] + bm[j]   (A = h^T, batched over b)
  {
    GemmArgs g = mk(h, mlpm_w, mlpm_b, nullptr, xm,
                    D, NPAT, SEQ,
                    /*rsA*/ 1, /*csA*/ D,
                    /*rsB*/ NPAT, /*csB*/ 1,
                    /*rsC*/ NPAT);
    g.nz = B;
    g.sAb = (long)SEQ * D;
    g.sCb = (long)D * NPAT;
    run_gemm(stream, g);
  }
  // 6) orix1[b,d] = xm[b,d,:] . ws + bs
  mlp_single_k<<<dim3((B * D + 255) / 256), dim3(256), 0, stream>>>(xm, mlps_w, mlps_b, orix);
  // 7) x1[p,b,d] = (xm[b,:,p]^T @ Wsfl + bsfl)[d] + orix1[b,d]   (batched over p)
  {
    GemmArgs g = mk(xm, sfl_w, sfl_b, orix, x1,
                    B, D, D,
                    /*rsA*/ D * NPAT, /*csA*/ NPAT,
                    /*rsB*/ D, /*csB*/ 1,
                    /*rsC*/ D);
    g.nz = NPAT;
    g.sAb = 1;                 // +p
    g.sCb = (long)B * D;       // x1[p]
    g.rsR = D; g.csR = 1; g.sRb = 0;  // orix broadcast over p
    run_gemm(stream, g);
  }
  // 8) single-token attention: softmax over 1 key == identity => o = v
  layernorm_k<<<dim3(ROWS_S), dim3(256), 0, stream>>>(x1, sn_g, sn_b, tmp, D);
  //    v = xn @ Wqkv[:, 1536:2304]
  run_gemm(stream, mk(tmp, sqkv_w + 2 * D, nullptr, nullptr, vbuf,
                      ROWS_S, D, D, D, 1, 3 * D, 1, D));
  //    xs = v @ Wproj + bproj + x1
  run_gemm(stream, mk(vbuf, sproj_w, sproj_b, x1, xs,
                      ROWS_S, D, D, D, 1, D, 1, D));
  // 9) head
  layernorm_k<<<dim3(ROWS_S), dim3(256), 0, stream>>>(xs, phln_g, phln_b, tmp, D);
  {
    GemmArgs g = mk(tmp, hf1_w, hf1_b, nullptr, hid,
                    ROWS_S, HIDN, D, D, 1, HIDN, 1, HIDN);
    g.act = 2;  // tanh
    run_gemm(stream, g);
  }
  run_gemm(stream, mk(hid, hf2_w, hf2_b, nullptr, out,
                      ROWS_S, OUTD, HIDN, HIDN, 1, OUTD, 1, OUTD));
}
